// BConvAttention2d_38079180046957
// MI455X (gfx1250) — 8-bit matrix operand order repaired and checked by reading, compiled, not run on hardware
//
#include <hip/hip_runtime.h>
#include <hip/hip_bf16.h>

typedef __attribute__((ext_vector_type(8))) int  v8i;
typedef __attribute__((ext_vector_type(4))) int  v4i;
typedef __attribute__((ext_vector_type(2))) int  v2i;
typedef __attribute__((ext_vector_type(4))) unsigned int v4u;

#define BATCH 32
#define CIN   64
#define COUT  64
#define IMH   128
#define IMW   128
#define BPACK_BYTES (9 * 4 * 32 * 32)
#define SFP_BYTES   (64 * 9 * 64)

#define WS_BP   0
#define WS_SFP  65536
#define WS_SXB  131072
#define WS_BSA  (131072 + 33554432)

__global__ __launch_bounds__(256) void pack_b_kernel(
    const float* __restrict__ of, signed char* __restrict__ Bp) {
  int i = blockIdx.x * 256 + threadIdx.x;
  if (i >= BPACK_BYTES) return;
  int byte = i & 3;
  int g    = (i >> 2) & 7;
  int lane = (i >> 5) & 31;
  int tj   = i >> 10;
  int t = tj >> 2, j = tj & 3;
  int u = t / 3, v = t % 3;
  int o = j * 16 + (lane & 15);
  int k = (g >> 1) * 16 + ((lane >= 16) ? 8 : 0) + (g & 1) * 4 + byte;
  float w = of[((o * CIN + k) * 3 + u) * 3 + v];
  Bp[i] = (w > 0.0f) ? (signed char)1 : ((w < 0.0f) ? (signed char)-1 : (signed char)0);
}

__global__ __launch_bounds__(256) void pack_sf_kernel(
    const float* __restrict__ pf, unsigned char* __restrict__ sfp) {
  int i = blockIdx.x * 256 + threadIdx.x;
  if (i >= SFP_BYTES) return;
  int c = i & 63;
  int t = (i >> 6) % 9;
  int p = i / (64 * 9);
  float w = pf[((size_t)c * 64 + p) * 9 + t];
  sfp[i] = (w < 0.0f) ? 1 : 0;
}

__global__ __launch_bounds__(256) void sign_kernel(
    const float* __restrict__ xin, unsigned char* __restrict__ sxb) {
  const int b  = blockIdx.x >> 7;
  const int y  = blockIdx.x & 127;
  const int c0 = (threadIdx.x & 15) * 4;
  const int x0 = threadIdx.x >> 4;
  const float* xrow = xin + (((size_t)b * CIN) << 14) + (size_t)y * IMW;
#pragma unroll
  for (int xi = 0; xi < 8; ++xi) {
    const int x = x0 + 16 * xi;
    unsigned int w = 0;
#pragma unroll
    for (int cc = 0; cc < 4; ++cc) {
      float xv = xrow[((size_t)(c0 + cc) << 14) + x];
      w |= (xv < 0.0f ? 1u : 0u) << (8 * cc);
    }
    *(unsigned int*)(sxb + ((((size_t)b * IMH + y) * IMW + x) * CIN + c0)) = w;
  }
}

__global__ __launch_bounds__(256) void selfconv_kernel(
    const unsigned char* __restrict__ sxb, const unsigned char* __restrict__ sfp,
    signed char* __restrict__ bsa) {
  __shared__ __align__(16) unsigned char sfl[4608];
  const int b = blockIdx.x >> 7;
  const int y = blockIdx.x & 127;
  const int prow = (y >> 4) << 3;
  {
    const int4* src = (const int4*)(sfp + (size_t)prow * 9 * 64);
    int4* dst = (int4*)sfl;
    int idx = threadIdx.x;
    if (idx < 288) dst[idx] = src[idx];
    idx += 256;
    if (idx < 288) dst[idx] = src[idx];
  }
  __syncthreads();

  const int x  = threadIdx.x >> 1;
  const int ch = (threadIdx.x & 1) * 32;
  const int px = x & 15, py = y & 15;
  const int pl = x >> 4;
  uint4 cnt0 = make_uint4(0, 0, 0, 0), cnt1 = make_uint4(0, 0, 0, 0);
  int nvalid = 0;
#pragma unroll
  for (int u = 0; u < 3; ++u) {
#pragma unroll
    for (int v = 0; v < 3; ++v) {
      const int ny = py + u - 1, nx = px + v - 1;
      if (ny >= 0 && ny < 16 && nx >= 0 && nx < 16) {
        ++nvalid;
        const uint4* sp = (const uint4*)(sxb +
            ((((size_t)b * IMH + (y + u - 1)) * IMW + (x + v - 1)) * CIN + ch));
        uint4 s0 = sp[0], s1 = sp[1];
        const uint4* fq = (const uint4*)(sfl + ((pl * 9 + (u * 3 + v)) * 64 + ch));
        uint4 f0 = fq[0], f1 = fq[1];
        cnt0.x += s0.x ^ f0.x; cnt0.y += s0.y ^ f0.y;
        cnt0.z += s0.z ^ f0.z; cnt0.w += s0.w ^ f0.w;
        cnt1.x += s1.x ^ f1.x; cnt1.y += s1.y ^ f1.y;
        cnt1.z += s1.z ^ f1.z; cnt1.w += s1.w ^ f1.w;
      }
    }
  }
  unsigned int cl[8] = {cnt0.x, cnt0.y, cnt0.z, cnt0.w, cnt1.x, cnt1.y, cnt1.z, cnt1.w};
  unsigned int ob[8];
#pragma unroll
  for (int d = 0; d < 8; ++d) {
    unsigned int c = cl[d], o = 0;
#pragma unroll
    for (int k = 0; k < 4; ++k) {
      int neg2 = (int)((c >> (8 * k)) & 255u) * 2;
      int r = (neg2 < nvalid) ? 1 : ((neg2 > nvalid) ? -1 : 0);
      o |= ((unsigned int)(r & 255)) << (8 * k);
    }
    ob[d] = o;
  }
  uint4* op = (uint4*)(bsa + ((((size_t)b * IMH + y) * IMW + x) * CIN + ch));
  op[0] = make_uint4(ob[0], ob[1], ob[2], ob[3]);
  op[1] = make_uint4(ob[4], ob[5], ob[6], ob[7]);
}

__global__ __launch_bounds__(256) void bconv_gemm_kernel(
    const signed char* __restrict__ bsa, const signed char* __restrict__ Bp,
    float* __restrict__ out) {
  __shared__ __align__(1024) unsigned char lds[65536];

  const int tid = threadIdx.x;

#if __has_builtin(__builtin_amdgcn_tensor_load_to_lds)
  if (tid < 32) {
    unsigned long long ga = (unsigned long long)Bp;
    unsigned int ldsa = (unsigned int)(size_t)(void*)lds;
    v4u g0;
    g0[0] = 1u;
    g0[1] = ldsa;
    g0[2] = (unsigned int)ga;
    g0[3] = ((unsigned int)(ga >> 32) & 0x01FFFFFFu) | 0x80000000u;
    v8i g1;
    g1[0] = (3 << 16);
    g1[1] = (int)((4608u & 0xFFFFu) << 16);
    g1[2] = (int)((4608u >> 16) | (1u << 16));
    g1[3] = (int)(4608u << 16);
    g1[4] = 0;
    g1[5] = 4608;
    g1[6] = 0; g1[7] = 0;
    v4i g2 = {}; v4i g3 = {};
#if defined(__clang_major__) && (__clang_major__ >= 23)
    v8i g1b = {};
    __builtin_amdgcn_tensor_load_to_lds(g0, g1, g2, g3, g1b, 0);
#else
    __builtin_amdgcn_tensor_load_to_lds(g0, g1, g2, g3, 0);
#endif
#if __has_builtin(__builtin_amdgcn_s_wait_tensorcnt)
    __builtin_amdgcn_s_wait_tensorcnt(0);
#endif
  }
  __syncthreads();
#endif

  {
    const int4* src = (const int4*)Bp;
    int4* dst = (int4*)lds;
#pragma unroll
    for (int i = 0; i < 9; ++i) dst[tid + 256 * i] = src[tid + 256 * i];
  }
  __syncthreads();

  const int lane  = tid & 31;
  const int wave  = tid >> 5;
  const int b     = blockIdx.x >> 6;
  const int y0    = (blockIdx.x & 63) * 2;
  const int r     = wave >> 2;
  const int y     = y0 + r;
  const int xBase = (wave & 3) * 32;
  const int row   = lane & 15;
  const int half8 = (lane >> 4) << 3;

  v8i acc[2][4];
#pragma unroll
  for (int s = 0; s < 2; ++s)
#pragma unroll
    for (int j = 0; j < 4; ++j) acc[s][j] = (v8i)(0);

#pragma unroll
  for (int t = 0; t < 9; ++t) {
    const int u = t / 3, v = t % 3;
    const int yy = y + u - 1;
    const int yc = (yy < 0) ? 0 : ((yy > IMH - 1) ? IMH - 1 : yy);
    const unsigned char* bt = lds + t * 4096 + lane * 32;
    v8i bm[4];
    bm[0] = *(const v8i*)(bt + 0);
    bm[1] = *(const v8i*)(bt + 1024);
    bm[2] = *(const v8i*)(bt + 2048);
    bm[3] = *(const v8i*)(bt + 3072);

#pragma unroll
    for (int s = 0; s < 2; ++s) {
      const int xx = xBase + s * 16 + row + v - 1;
      const int xc = (xx < 0) ? 0 : ((xx > IMW - 1) ? IMW - 1 : xx);
      const int msk = ((yy == yc) && (xx == xc)) ? -1 : 0;
      const int* p = (const int*)(bsa +
          ((((size_t)b * IMH + yc) * IMW + xc) * CIN + half8));
      v2i q0 = *(const v2i*)(p + 0);
      v2i q1 = *(const v2i*)(p + 4);
      v2i q2 = *(const v2i*)(p + 8);
      v2i q3 = *(const v2i*)(p + 12);
      v8i a;
      a[0] = q0[0] & msk; a[1] = q0[1] & msk;
      a[2] = q1[0] & msk; a[3] = q1[1] & msk;
      a[4] = q2[0] & msk; a[5] = q2[1] & msk;
      a[6] = q3[0] & msk; a[7] = q3[1] & msk;
      acc[s][0] = __builtin_amdgcn_wmma_i32_16x16x64_iu8(true, a, true, bm[0], acc[s][0], false, false);
      acc[s][1] = __builtin_amdgcn_wmma_i32_16x16x64_iu8(true, a, true, bm[1], acc[s][1], false, false);
      acc[s][2] = __builtin_amdgcn_wmma_i32_16x16x64_iu8(true, a, true, bm[2], acc[s][2], false, false);
      acc[s][3] = __builtin_amdgcn_wmma_i32_16x16x64_iu8(true, a, true, bm[3], acc[s][3], false, false);
    }
  }

  __syncthreads();
  float* stage = (float*)lds;
#pragma unroll
  for (int s = 0; s < 2; ++s) {
#pragma unroll
    for (int g = 0; g < 8; ++g) {
      const int xo = xBase + s * 16 + g + half8;
      float* st = stage + (size_t)r * COUT * IMW + xo;
      st[(row +  0) * IMW] = (float)acc[s][0][g];
      st[(row + 16) * IMW] = (float)acc[s][1][g];
      st[(row + 32) * IMW] = (float)acc[s][2][g];
      st[(row + 48) * IMW] = (float)acc[s][3][g];
    }
  }
  __syncthreads();

#pragma unroll
  for (int it = 0; it < 16; ++it) {
    const int f4 = tid + 256 * it;
    const int i  = f4 << 2;
    const int rr = i >> 13;
    const int o  = (i >> 7) & 63;
    const int x  = i & 127;
    *(float4*)(out + ((((size_t)b * COUT + o) * IMH) + (y0 + rr)) * IMW + x) =
        *(const float4*)(stage + i);
  }
}

extern "C" void kernel_launch(void* const* d_in, const int* in_sizes, int n_in,
                              void* d_out, int out_size, void* d_ws, size_t ws_size,
                              hipStream_t stream) {
  const float* input = (const float*)d_in[0];
  const float* patch_filters  = (const float*)d_in[3];
  const float* output_filters = (const float*)d_in[4];

  signed char*  Bp  = (signed char*)d_ws + WS_BP;
  unsigned char* sfp = (unsigned char*)d_ws + WS_SFP;
  unsigned char* sxb = (unsigned char*)d_ws + WS_SXB;
  signed char*  bsa = (signed char*)d_ws + WS_BSA;

  pack_b_kernel<<<(BPACK_BYTES + 255) / 256, 256, 0, stream>>>(output_filters, Bp);
  pack_sf_kernel<<<(SFP_BYTES + 255) / 256, 256, 0, stream>>>(patch_filters, sfp);
  sign_kernel<<<BATCH * IMH, 256, 0, stream>>>(input, sxb);
  selfconv_kernel<<<BATCH * IMH, 256, 0, stream>>>(sxb, sfp, bsa);
  bconv_gemm_kernel<<<BATCH * (IMH / 2), 256, 0, stream>>>(bsa, Bp, (float*)d_out);
}
